// SliceAttention_42099269436310
// MI455X (gfx1250) — compile-verified, matrix operand order checked by reading, not run on hardware
//
#include <hip/hip_runtime.h>
#include <hip/hip_bf16.h>

// Problem constants (reference: b=4, n=8192, H=8, M=32, D=128, HID=1024)
#define BB   4
#define NN   8192
#define HHD  8            // heads
#define MMS  32           // slices
#define DDH  128          // head dim
#define HID  1024
#define TOK  (BB * NN)    // 32768 tokens

typedef __bf16 bf16;
typedef __attribute__((ext_vector_type(16))) bf16  v16bf;
typedef __attribute__((ext_vector_type(8)))  bf16  v8bf;
typedef __attribute__((ext_vector_type(8)))  float v8f;

// ---------------------------------------------------------------------------
// WMMA fragment loaders (wave32, V_WMMA_F32_16X16X32_BF16 layouts per ISA §7.12.2)
//   A 16x32 bf16: lane L -> row M = L&15; element i -> K = (i>>3)*16 + ((L>>4)<<3) + (i&7)
//   B^T-contiguous loads are symmetric with N = L&15.
// ---------------------------------------------------------------------------
__device__ __forceinline__ v16bf load_frag_contig(const bf16* __restrict__ base,
                                                  long long stride, int row_base, int k0) {
  const int lane = threadIdx.x & 31;
  const bf16* p = base + (long long)(row_base + (lane & 15)) * stride
                       + (k0 + ((lane >> 4) << 3));
  v8bf lo = *(const v8bf*)p;          // K = kb .. kb+7
  v8bf hi = *(const v8bf*)(p + 16);   // K = kb+16 .. kb+23
  return __builtin_shufflevector(lo, hi, 0,1,2,3,4,5,6,7,8,9,10,11,12,13,14,15);
}

// B stored row-major [K][N] (strided gather along K)
__device__ __forceinline__ v16bf load_fragB_rowmajor(const bf16* __restrict__ base,
                                                     long long ldb, int n_base, int k0) {
  const int lane = threadIdx.x & 31;
  const bf16* p = base + (long long)(k0 + ((lane >> 4) << 3)) * ldb + (n_base + (lane & 15));
  v16bf f;
#pragma unroll
  for (int i = 0; i < 8; ++i) {
    f[i]     = p[(long long)i * ldb];
    f[i + 8] = p[(long long)(i + 16) * ldb];
  }
  return f;
}

// ---------------------------------------------------------------------------
// Generic batched bf16 GEMM, f32 accumulate.  C = A[MxK] * B + bias
//   B_IS_NT=1 : B passed as [N][K] (contiguous fragment loads)
//   B_IS_NT=0 : B passed as [K][N] row-major
// Register blocking: one wave owns a 32x64 macro-tile = 2 A-frags x 4 B-frags
// -> 8 v_wmma per k-step over 6 fragment loads. Block = 256 thr = 8 waves.
// __launch_bounds__(256,1): allow a large VGPR budget so the 8 f32 accumulators
// (64 VGPRs) + 6 bf16 fragments (48 VGPRs) stay resident -- no scratch spills.
// Batch z decomposed as z = zo*innerCount + zi with independent offsets.
// Requires M % 32 == 0, N % 64 == 0, K % 32 == 0 (all shapes here comply).
// ---------------------------------------------------------------------------
template<bool B_IS_NT, bool OUT_F32, bool OUT_BF16, bool HAS_BIAS>
__global__ void __launch_bounds__(256, 1)
k_gemm(const bf16* __restrict__ A, long long lda,
       const bf16* __restrict__ B, long long ldb,
       float* __restrict__ Cf, bf16* __restrict__ Cb, long long ldc,
       const float* __restrict__ bias,
       int M, int N, int K,
       long long aOffO, long long aOffI,
       long long bOffO, long long bOffI,
       long long cOffO, long long cOffI, int innerCount) {
  const int z  = blockIdx.z;
  const int zo = z / innerCount, zi = z % innerCount;
  A += (long long)zo * aOffO + (long long)zi * aOffI;
  B += (long long)zo * bOffO + (long long)zi * bOffI;
  const long long coff = (long long)zo * cOffO + (long long)zi * cOffI;

  const int wave     = threadIdx.x >> 5;
  const int tilesN64 = N >> 6;                          // macro-tiles along N
  const long long tile  = (long long)blockIdx.x * (blockDim.x >> 5) + wave;
  const long long total = (long long)(M >> 5) * tilesN64;
  if (tile >= total) return;                 // uniform per wave: EXEC stays all-ones
  const int tm = (int)(tile / tilesN64) << 5;           // 32 rows
  const int tn = (int)(tile % tilesN64) << 6;           // 64 cols

  v8f acc[2][4] = {};
  for (int k0 = 0; k0 < K; k0 += 32) {
    v16bf a0 = load_frag_contig(A, lda, tm,      k0);
    v16bf a1 = load_frag_contig(A, lda, tm + 16, k0);
    v16bf b[4];
#pragma unroll
    for (int j = 0; j < 4; ++j) {
      if constexpr (B_IS_NT) b[j] = load_frag_contig(B, ldb, tn + 16 * j, k0);
      else                   b[j] = load_fragB_rowmajor(B, ldb, tn + 16 * j, k0);
    }
#pragma unroll
    for (int j = 0; j < 4; ++j) {
      acc[0][j] = __builtin_amdgcn_wmma_f32_16x16x32_bf16(false, a0, false, b[j],
                                                          (short)0, acc[0][j], false, false);
      acc[1][j] = __builtin_amdgcn_wmma_f32_16x16x32_bf16(false, a1, false, b[j],
                                                          (short)0, acc[1][j], false, false);
    }
    if (k0 + 32 < K) {                       // CDNA5 global_prefetch_b8 of next K-slab
      const int lane = threadIdx.x & 31;
      __builtin_prefetch(A + (long long)(tm + (lane & 15)) * lda
                           + (k0 + 32) + ((lane >> 4) << 3), 0, 1);
      if constexpr (B_IS_NT)
        __builtin_prefetch(B + (long long)(tn + (lane & 15)) * ldb
                             + (k0 + 32) + ((lane >> 4) << 3), 0, 1);
    }
  }

  // C/D layout: lane L -> col N = L&15 ; VGPR r -> row M = ((L>>4)<<3) + r
  const int lane = threadIdx.x & 31;
#pragma unroll
  for (int i = 0; i < 2; ++i) {
    const int rbase = tm + 16 * i + ((lane >> 4) << 3);
#pragma unroll
    for (int j = 0; j < 4; ++j) {
      const int col = tn + 16 * j + (lane & 15);
      float bv = 0.f;
      if constexpr (HAS_BIAS) bv = bias[col];
#pragma unroll
      for (int r = 0; r < 8; ++r) {
        const float v = acc[i][j][r] + bv;
        const long long idx = coff + (long long)(rbase + r) * ldc + col;
        if constexpr (OUT_F32)  Cf[idx] = v;
        if constexpr (OUT_BF16) Cb[idx] = (bf16)v;
      }
    }
  }
}

// ---------------------------------------------------------------------------
// Small helper kernels
// ---------------------------------------------------------------------------
__global__ void k_to_bf16(const float* __restrict__ src, bf16* __restrict__ dst, long long n) {
  long long i = (long long)blockIdx.x * blockDim.x + threadIdx.x;
  if (i < n) dst[i] = (bf16)src[i];
}

// src f32 [K][N] -> dst bf16 [N][K]
__global__ void k_transpose_bf16(const float* __restrict__ src, bf16* __restrict__ dst,
                                 int K, int N) {
  long long i = (long long)blockIdx.x * blockDim.x + threadIdx.x;
  if (i >= (long long)K * N) return;
  const int k = (int)(i / N), n = (int)(i % N);
  dst[(long long)n * K + k] = (bf16)src[i];
}

__global__ void k_zero(float* __restrict__ p, int n) {
  int i = blockIdx.x * blockDim.x + threadIdx.x;
  if (i < n) p[i] = 0.f;
}

__global__ void k_silu(const float* __restrict__ c, float* __restrict__ sc, int n) {
  int i = blockIdx.x * blockDim.x + threadIdx.x;
  if (i < n) { float x = c[i]; sc[i] = x / (1.f + __expf(-x)); }
}

// xq[b][j] = silu(c[b]) . Wq[:,j] + bq[j] ; j linearizes (h,m,d) exactly as stored
__global__ void k_xq(const float* __restrict__ sc, const float* __restrict__ Wq,
                     const float* __restrict__ bq, bf16* __restrict__ xq) {
  const int j = blockIdx.x * blockDim.x + threadIdx.x;   // 0..32767
  const int b = blockIdx.y;
  const float* s = sc + b * HID;
  float acc = bq[j];
  for (int k = 0; k < HID; ++k) acc += s[k] * Wq[(long long)k * (HHD*MMS*DDH) + j];
  xq[(long long)b * (HHD*MMS*DDH) + j] = (bf16)acc;
}

// softmax over slice axis m (32 values) per (b,h,n); emits bf16 W, W^T, slice_norm
__global__ void k_softmax(const float* __restrict__ scores,   // [bh][32][8192]
                          const float* __restrict__ alpha,    // [8]
                          bf16* __restrict__ W,               // [bh][32][8192]
                          bf16* __restrict__ WT,              // [bh][8192][32]
                          float* __restrict__ norm) {         // [bh][32]
  const int bh = blockIdx.y;
  const int h  = bh & (HHD - 1);
  const int n  = blockIdx.x * blockDim.x + threadIdx.x;
  const float a = alpha[h];
  const float* s = scores + (long long)bh * MMS * NN + n;

  float v[MMS];
  float mx = -1e30f;
#pragma unroll
  for (int m = 0; m < MMS; ++m) { v[m] = s[(long long)m * NN] * a; mx = fmaxf(mx, v[m]); }
  float sum = 0.f;
#pragma unroll
  for (int m = 0; m < MMS; ++m) { v[m] = __expf(v[m] - mx); sum += v[m]; }
  const float inv = 1.f / sum;

  __shared__ float lnorm[MMS];
  if (threadIdx.x < MMS) lnorm[threadIdx.x] = 0.f;
  __syncthreads();

  bf16 wt[MMS];
#pragma unroll
  for (int m = 0; m < MMS; ++m) {
    const float w = v[m] * inv;
    atomicAdd(&lnorm[m], w);
    W[(long long)bh * MMS * NN + (long long)m * NN + n] = (bf16)w;
    wt[m] = (bf16)w;
  }
  // W^T row for token n: 64B contiguous -> 4 x 16B stores
  v8bf* dst = (v8bf*)(WT + (long long)bh * NN * MMS + (long long)n * MMS);
#pragma unroll
  for (int g = 0; g < 4; ++g) {
    v8bf pk;
#pragma unroll
    for (int e = 0; e < 8; ++e) pk[e] = wt[g * 8 + e];
    dst[g] = pk;
  }
  __syncthreads();
  if (threadIdx.x < MMS) atomicAdd(&norm[bh * MMS + threadIdx.x], lnorm[threadIdx.x]);
}

// per-(b,h): slice_norm divide -> qkv projection -> 32x32 inner MHA -> out_token^T (bf16)
__global__ void k_middle(const float* __restrict__ st_raw,   // [bh][32][128]
                         const float* __restrict__ norm,     // [bh][32]
                         const float* __restrict__ qkv_proj, // [8][128][384]
                         bf16* __restrict__ outTT) {         // [bh][128][32]
  __shared__ float st[MMS * DDH];      // 16 KB (later aliased by attn)
  __shared__ float qkv[MMS * 3 * DDH]; // 48 KB
  const int bh = blockIdx.x;
  const int h  = bh & (HHD - 1);

  for (int i = threadIdx.x; i < MMS * DDH; i += blockDim.x) {
    const int m = i >> 7;
    st[i] = st_raw[(long long)bh * MMS * DDH + i] / (norm[bh * MMS + m] + 1e-5f);
  }
  __syncthreads();

  const float* P = qkv_proj + (long long)h * DDH * (3 * DDH);
  for (int i = threadIdx.x; i < MMS * 3 * DDH; i += blockDim.x) {
    const int m = i / (3 * DDH), e = i % (3 * DDH);
    float s = 0.f;
    for (int d = 0; d < DDH; ++d) s += st[m * DDH + d] * P[d * (3 * DDH) + e];
    qkv[i] = s;
  }
  __syncthreads();

  float* attn = st;  // alias: st no longer needed (1024 floats of the 4096)
  for (int i = threadIdx.x; i < MMS * MMS; i += blockDim.x) {
    const int qi = i >> 5, ki = i & 31;
    float s = 0.f;
    for (int d = 0; d < DDH; ++d)
      s += qkv[qi * 384 + d] * qkv[ki * 384 + DDH + d];
    attn[i] = s * 0.08838834764831845f;   // D^-0.5
  }
  __syncthreads();

  for (int qi = threadIdx.x; qi < MMS; qi += blockDim.x) {
    float mx = -1e30f;
    for (int k = 0; k < MMS; ++k) mx = fmaxf(mx, attn[qi * 32 + k]);
    float sum = 0.f;
    for (int k = 0; k < MMS; ++k) { float e = __expf(attn[qi * 32 + k] - mx); attn[qi * 32 + k] = e; sum += e; }
    const float inv = 1.f / sum;
    for (int k = 0; k < MMS; ++k) attn[qi * 32 + k] *= inv;
  }
  __syncthreads();

  for (int i = threadIdx.x; i < DDH * MMS; i += blockDim.x) {
    const int d = i >> 5, m = i & 31;
    float s = 0.f;
    for (int k = 0; k < MMS; ++k) s += attn[m * 32 + k] * qkv[k * 384 + 2 * DDH + d];
    outTT[(long long)bh * DDH * MMS + i] = (bf16)s;  // [d][m]
  }
}

// ---------------------------------------------------------------------------
extern "C" void kernel_launch(void* const* d_in, const int* in_sizes, int n_in,
                              void* d_out, int out_size, void* d_ws, size_t ws_size,
                              hipStream_t stream) {
  const float* x      = (const float*)d_in[0];
  const float* c      = (const float*)d_in[1];
  const float* wkv_w  = (const float*)d_in[2];
  const float* wkv_b  = (const float*)d_in[3];
  const float* wq_w   = (const float*)d_in[4];
  const float* wq_b   = (const float*)d_in[5];
  const float* alpha  = (const float*)d_in[6];
  const float* qkvp   = (const float*)d_in[7];
  const float* out_w  = (const float*)d_in[8];
  const float* out_b  = (const float*)d_in[9];
  float* out          = (float*)d_out;

  // workspace carve-out (256B aligned), ~343 MB total
  char* base = (char*)d_ws;
  size_t off = 0;
  auto alloc = [&](size_t bytes) -> char* {
    char* p = base + off;
    off = (off + bytes + 255) & ~(size_t)255;
    return p;
  };
  bf16*  x_bf   = (bf16*) alloc((size_t)TOK * HID * 2);            // 64 MB
  bf16*  wkvT   = (bf16*) alloc((size_t)(2*HID) * HID * 2);        // 4 MB  [2048][1024]
  bf16*  outwT  = (bf16*) alloc((size_t)HID * HID * 2);            // 2 MB  [1024][1024]
  bf16*  kv_bf  = (bf16*) alloc((size_t)TOK * 2*HID * 2);          // 128 MB [tok][2048]
  float* sc     = (float*)alloc((size_t)BB * HID * 4);             // silu(c)
  bf16*  xq_bf  = (bf16*) alloc((size_t)BB * HHD*MMS*DDH * 2);     // [b][h][m][d]
  float* scores = (float*)alloc((size_t)BB*HHD*MMS*NN * 4);        // 32 MB
  bf16*  W_bf   = (bf16*) alloc((size_t)BB*HHD*MMS*NN * 2);        // 16 MB
  bf16*  WT_bf  = (bf16*) alloc((size_t)BB*HHD*NN*MMS * 2);        // 16 MB
  float* snorm  = (float*)alloc((size_t)BB*HHD*MMS * 4);
  float* st_raw = (float*)alloc((size_t)BB*HHD*MMS*DDH * 4);
  bf16*  outTT  = (bf16*) alloc((size_t)BB*HHD*DDH*MMS * 2);
  bf16*  hid    = (bf16*) alloc((size_t)TOK * HID * 2);            // 64 MB

  // 0) weight prep + init
  k_transpose_bf16<<<(2*HID*HID + 255)/256, 256, 0, stream>>>(wkv_w, wkvT, HID, 2*HID);
  k_transpose_bf16<<<(HID*HID + 255)/256, 256, 0, stream>>>(out_w, outwT, HID, HID);
  k_zero<<<(BB*HHD*MMS + 255)/256, 256, 0, stream>>>(snorm, BB*HHD*MMS);
  k_to_bf16<<<(int)(((size_t)TOK*HID + 255)/256), 256, 0, stream>>>(x, x_bf, (long long)TOK*HID);

  // 1) kv = x @ Wkv + b  (bf16 out, [tok][2048])
  {
    long long tiles = (long long)(TOK/32) * (2*HID/64);
    k_gemm<true,false,true,true><<<dim3((unsigned)((tiles+7)/8),1,1), 256, 0, stream>>>(
        x_bf, HID, wkvT, HID, nullptr, kv_bf, 2*HID, wkv_b,
        TOK, 2*HID, HID, 0,0, 0,0, 0,0, 1);
  }

  // 2) xq = silu(c) @ Wq + b -> bf16 [b][h][m][d]
  k_silu<<<(BB*HID + 255)/256, 256, 0, stream>>>(c, sc, BB*HID);
  k_xq<<<dim3(HHD*MMS*DDH/256, BB), 256, 0, stream>>>(sc, wq_w, wq_b, xq_bf);

  // 3) scores[bh][m][n] = xq[bh] @ xk[bh]^T  (xk = kv[:, h*128 : h*128+128])
  {
    long long tiles = (long long)(MMS/32) * (NN/64);  // 128
    k_gemm<true,true,false,false><<<dim3((unsigned)((tiles+7)/8),1,BB*HHD), 256, 0, stream>>>(
        xq_bf, DDH, kv_bf, 2*HID, scores, nullptr, NN, nullptr,
        MMS, NN, DDH,
        (long long)HHD*MMS*DDH, (long long)MMS*DDH,          // A: b,h
        (long long)NN*2*HID,    DDH,                          // B: b outer, h*128 inner
        (long long)HHD*MMS*NN,  (long long)MMS*NN, HHD);      // C
  }

  // 4) softmax over m + slice_norm + bf16 W / W^T
  k_softmax<<<dim3(NN/256, BB*HHD), 256, 0, stream>>>(scores, alpha, W_bf, WT_bf, snorm);

  // 5) slice_token_raw[bh][m][d] = W[bh] @ xv[bh]   (xv rows strided inside kv)
  {
    long long tiles = (long long)(MMS/32) * (DDH/64);  // 2
    k_gemm<false,true,false,false><<<dim3((unsigned)((tiles+7)/8),1,BB*HHD), 256, 0, stream>>>(
        W_bf, NN, kv_bf + HID, 2*HID, st_raw, nullptr, DDH, nullptr,
        MMS, DDH, NN,
        (long long)HHD*MMS*NN,  (long long)MMS*NN,
        (long long)NN*2*HID,    DDH,
        (long long)HHD*MMS*DDH, (long long)MMS*DDH, HHD);
  }

  // 6) normalize, per-head qkv proj, 32x32 inner MHA -> out_token^T bf16
  k_middle<<<BB*HHD, 128, 0, stream>>>(st_raw, snorm, qkvp, outTT);

  // 7) scatter: hid[b][n][h*128+d] = sum_m WT[bh][n][m] * outTT[bh][d][m]  (K=32)
  {
    long long tiles = (long long)(NN/32) * (DDH/64);  // 512
    k_gemm<true,false,true,false><<<dim3((unsigned)((tiles+7)/8),1,BB*HHD), 256, 0, stream>>>(
        WT_bf, MMS, outTT, MMS, nullptr, hid, HID, nullptr,
        NN, DDH, MMS,
        (long long)HHD*NN*MMS,  (long long)NN*MMS,
        (long long)HHD*DDH*MMS, (long long)DDH*MMS,
        (long long)NN*HID,      DDH, HHD);              // C: b outer, h*128 inner
  }

  // 8) out = hid @ W_out + b  (f32 to d_out)
  {
    long long tiles = (long long)(TOK/32) * (HID/64);
    k_gemm<true,true,false,true><<<dim3((unsigned)((tiles+7)/8),1,1), 256, 0, stream>>>(
        hid, HID, outwT, HID, out, nullptr, HID, out_b,
        TOK, HID, HID, 0,0, 0,0, 0,0, 1);
  }
}
